// STGA_47545287967260
// MI455X (gfx1250) — matrix operand order repaired and checked by reading, not compiled, not run on hardware
//
#include <hip/hip_runtime.h>

// ---------------- problem constants ----------------
constexpr int Bn  = 8;      // batch
constexpr int Tn  = 12;     // time steps
constexpr int Sn  = 1000;   // stations
constexpr int DHD = 8, DST = 8;
constexpr int Fn  = 16;     // DHD + DST
constexpr int Hn  = 4, Cn = 16;
constexpr int HC  = 64;     // H*C
constexpr int HL  = 128;
constexpr int OUTn = 16;
constexpr int En  = 16000;
constexpr int Etn = En + Sn;          // + self loops
constexpr int Gn  = Bn * Tn;          // 96 graphs
constexpr int BSn = Bn * Sn;          // 8000 sequences
constexpr int KC  = HC + HL;          // 192 combined LSTM K

// ---------------- workspace layout (bytes) ----------------
constexpr size_t OFF_XW  = 0;                              // f32 [G*S][64]
constexpr size_t OFF_AS  = OFF_XW  + (size_t)Gn*Sn*HC*4;   // f32 [G*S][H]
constexpr size_t OFF_AD  = OFF_AS  + (size_t)Gn*Sn*Hn*4;
constexpr size_t OFF_XBF = OFF_AD  + (size_t)Gn*Sn*Hn*4;   // bf16 [T][BS][64]
constexpr size_t OFF_WC  = OFF_XBF + (size_t)Tn*BSn*HC*2;  // bf16 [512][192]
constexpr size_t OFF_BC  = OFF_WC  + (size_t)4*HL*KC*2;    // f32  [512]
constexpr size_t OFF_HBF = OFF_BC  + (size_t)4*HL*4;       // bf16 [BS][128]
constexpr size_t OFF_CB  = OFF_HBF + (size_t)BSn*HL*2;     // f32  [BS][128]

// ---------------- WMMA types ----------------
typedef __bf16 v16bf __attribute__((ext_vector_type(16)));
typedef float  v8f   __attribute__((ext_vector_type(8)));

union FragBF {                 // 32-byte WMMA A/B operand
  v16bf v;
  uint4 q[2];
  unsigned short us[16];
};

// ---------------- helpers ----------------
__device__ __forceinline__ unsigned short f2bf(float f) {
  unsigned u = __builtin_bit_cast(unsigned, f);
  u += 0x7FFFu + ((u >> 16) & 1u);               // round-to-nearest-even
  return (unsigned short)(u >> 16);
}
__device__ __forceinline__ float bf2f(unsigned short s) {
  return __builtin_bit_cast(float, (unsigned)s << 16);
}
// order-preserving float <-> uint for atomicMax
__device__ __forceinline__ unsigned fenc(float f) {
  int i = __builtin_bit_cast(int, f);
  unsigned u = (unsigned)i;
  return (i < 0) ? ~u : (u | 0x80000000u);
}
__device__ __forceinline__ float fdec(unsigned u) {
  unsigned v = (u & 0x80000000u) ? (u & 0x7FFFFFFFu) : ~u;
  return __builtin_bit_cast(float, v);
}
__device__ __forceinline__ float sigm(float x) { return 1.0f / (1.0f + __expf(-x)); }

// =====================================================================
// K0: pack LSTM weights to bf16 [512][192], combine biases, zero h0/c0
//     (split loops -> contiguous unpredicated streaming loads)
// =====================================================================
__global__ __launch_bounds__(256) void k_prep(
    const float* __restrict__ Wih, const float* __restrict__ Whh,
    const float* __restrict__ bih, const float* __restrict__ bhh,
    unsigned short* __restrict__ Wc, float* __restrict__ bc,
    unsigned short* __restrict__ hbf, float* __restrict__ cbuf) {
  const int stride = gridDim.x * blockDim.x;
  const int g0 = blockIdx.x * blockDim.x + threadIdx.x;
  for (int i = g0; i < 4 * HL * HC; i += stride) {       // W_ih -> cols 0..63
    int n = i >> 6, k = i & 63;
    Wc[n * KC + k] = f2bf(Wih[i]);
  }
  for (int i = g0; i < 4 * HL * HL; i += stride) {       // W_hh -> cols 64..191
    int n = i >> 7, k = i & 127;
    Wc[n * KC + HC + k] = f2bf(Whh[i]);
  }
  for (int i = g0; i < 4 * HL; i += stride) bc[i] = bih[i] + bhh[i];
  for (int i = g0; i < BSn * HL; i += stride) { hbf[i] = 0; cbuf[i] = 0.0f; }
}

// =====================================================================
// K1: GAT linear  xw[G*S,64] = x[G*S,16] @ W_gat^T   (bf16 WMMA, K=16 pad 32)
//   one wave -> 16 rows x 64 cols (4 WMMAs); all fragment loads are
//   unconditional float4 -> b128; zero padding applied by value masks.
// =====================================================================
__global__ __launch_bounds__(128) void k_gat_linear(
    const float* __restrict__ hist, const float* __restrict__ stat,
    const float* __restrict__ Wg, float* __restrict__ xw) {
  const int wave = blockIdx.x * 4 + (threadIdx.x >> 5);
  const int lane = threadIdx.x & 31;
  const int lane16 = lane & 15;
  const int hi = lane >> 4;
  const int tile = wave;                       // 6000 tiles of 16 rows

  // ---- A fragment: row = tile*16 + lane16; lanes<16 hold K0..7 (hist),
  //      lanes>=16 hold K8..15 (static); K16..31 zero-padded ----
  const int node = tile * 16 + lane16;         // flat (g,s)
  const int g = node / Sn, s = node - g * Sn;
  const int b = g / Tn;
  const float* ap = hi ? (stat + ((size_t)b * Sn + s) * DST)
                       : (hist + (size_t)node * DHD);
  float4 a0 = ((const float4*)ap)[0];
  float4 a1 = ((const float4*)ap)[1];
  const float av[8] = {a0.x, a0.y, a0.z, a0.w, a1.x, a1.y, a1.z, a1.w};
  FragBF A;
#pragma unroll
  for (int i = 0; i < 8; ++i) A.us[i] = f2bf(av[i]);
#pragma unroll
  for (int i = 8; i < 16; ++i) A.us[i] = 0;

  const unsigned short bmask = hi ? (unsigned short)0 : (unsigned short)0xFFFF;

#pragma unroll
  for (int nt = 0; nt < 4; ++nt) {
    const int n0 = nt * 16;
    // B fragment: column n0+lane16, elements 0..7 = W_gat[n][hi*8..hi*8+7];
    // elements 8..15 hold K16..31 which are zero.
    const float4* wp4 = (const float4*)(Wg + (size_t)(n0 + lane16) * Fn + hi * 8);
    float4 w0 = wp4[0], w1 = wp4[1];
    const float wv[8] = {w0.x, w0.y, w0.z, w0.w, w1.x, w1.y, w1.z, w1.w};
    FragBF Bf;
#pragma unroll
    for (int i = 0; i < 16; ++i)
      Bf.us[i] = (i < 8) ? f2bf(wv[i & 7]) : (unsigned short)0;
    v8f acc = {};
    acc = __builtin_amdgcn_wmma_f32_16x16x32_bf16(false, A.v, false, Bf.v,
                                                  (short)0, acc, false, false);
#pragma unroll
    for (int r = 0; r < 8; ++r) {
      int M = r + hi * 8;
      xw[(size_t)(tile * 16 + M) * HC + n0 + lane16] = acc[r];
    }
  }
}

// =====================================================================
// K2: per-node attention coefficients a_s, a_d  [G*S][H]
// =====================================================================
__global__ __launch_bounds__(256) void k_att(
    const float* __restrict__ xw, const float* __restrict__ att_s,
    const float* __restrict__ att_d, float* __restrict__ a_s,
    float* __restrict__ a_d) {
  int idx = blockIdx.x * blockDim.x + threadIdx.x;   // G*S*H
  if (idx >= Gn * Sn * Hn) return;
  int n = idx / Hn, h = idx % Hn;
  const float4* xp = (const float4*)(xw + (size_t)n * HC + h * Cn);
  const float4* as4 = (const float4*)(att_s + h * Cn);
  const float4* ad4 = (const float4*)(att_d + h * Cn);
  float ss = 0.f, sd = 0.f;
#pragma unroll
  for (int q = 0; q < 4; ++q) {
    float4 v = xp[q], s4 = as4[q], d4 = ad4[q];
    ss += v.x * s4.x + v.y * s4.y + v.z * s4.z + v.w * s4.w;
    sd += v.x * d4.x + v.y * d4.y + v.z * d4.z + v.w * d4.w;
  }
  a_s[idx] = ss;
  a_d[idx] = sd;
}

// =====================================================================
// K3: per-graph edge softmax + aggregation, fully in LDS (288 KB / WGP)
//   out -> bf16 gat features laid out [t][b*S+s][64] for the LSTM
// =====================================================================
__global__ __launch_bounds__(1024) void k_graph(
    const int* __restrict__ ei, const float* __restrict__ a_s,
    const float* __restrict__ a_d, const float* __restrict__ xw,
    const float* __restrict__ bgat, unsigned short* __restrict__ xbf) {
  extern __shared__ __align__(16) char smem[];
  unsigned* mEnc = (unsigned*)smem;                    // [S*H]  16000 B
  float* den = (float*)(smem + Sn * Hn * 4);           // [S*H]  16000 B
  float* agg = (float*)(smem + 2 * Sn * Hn * 4);       // [S*64] 256000 B

  const int g = blockIdx.x;
  const int b = g / Tn, t = g - b * Tn;
  const int tid = threadIdx.x, nthr = blockDim.x;
  const float* asg = a_s + (size_t)g * Sn * Hn;
  const float* adg = a_d + (size_t)g * Sn * Hn;
  const float* xwg = xw + (size_t)g * Sn * HC;

  for (int i = tid; i < Sn * Hn; i += nthr) { mEnc[i] = 0u; den[i] = 0.f; }
  for (int i = tid; i < Sn * HC; i += nthr) agg[i] = 0.f;
  __syncthreads();

  // pass 1: segment max of leaky-relu logits over destination nodes
  for (int idx = tid; idx < Etn * Hn; idx += nthr) {
    int e = idx >> 2, h = idx & 3;
    int src = (e < En) ? ei[e] : (e - En);
    int dst = (e < En) ? ei[En + e] : (e - En);
    float v = asg[src * Hn + h] + adg[dst * Hn + h];
    float l = (v > 0.f) ? v : 0.2f * v;
    atomicMax(&mEnc[dst * Hn + h], fenc(l));
  }
  __syncthreads();

  // pass 2: exp, denominator, weighted message accumulation
  for (int idx = tid; idx < Etn * Hn; idx += nthr) {
    int e = idx >> 2, h = idx & 3;
    int src = (e < En) ? ei[e] : (e - En);
    int dst = (e < En) ? ei[En + e] : (e - En);
    float v = asg[src * Hn + h] + adg[dst * Hn + h];
    float l = (v > 0.f) ? v : 0.2f * v;
    float ex = __expf(l - fdec(mEnc[dst * Hn + h]));
    atomicAdd(&den[dst * Hn + h], ex);
    const float4* xp = (const float4*)(xwg + (size_t)src * HC + h * Cn);
    float* ag = agg + dst * HC + h * Cn;
#pragma unroll
    for (int q = 0; q < 4; ++q) {
      float4 m = xp[q];
      atomicAdd(&ag[q * 4 + 0], ex * m.x);
      atomicAdd(&ag[q * 4 + 1], ex * m.y);
      atomicAdd(&ag[q * 4 + 2], ex * m.z);
      atomicAdd(&ag[q * 4 + 3], ex * m.w);
    }
  }
  __syncthreads();

  // normalize, add bias, emit bf16 in LSTM layout [t][b*S+s][64]
  for (int i = tid; i < Sn * HC; i += nthr) {
    int s = i >> 6, j = i & 63, h = j >> 4;
    float o = agg[i] / den[s * Hn + h] + bgat[j];
    xbf[((size_t)t * BSn + (size_t)b * Sn + s) * HC + j] = f2bf(o);
  }
}

// =====================================================================
// K4: one LSTM time step.  z[16,512] per block via bf16 WMMA (K=192),
//     gate nonlinearity in-block.  12 stream-ordered launches.
//     A staging split into two contiguous dword copies (xt tile, h tile).
// =====================================================================
__global__ __launch_bounds__(256) void k_lstm_step(
    int t, const unsigned short* __restrict__ xbf,
    const unsigned short* __restrict__ Wc, const float* __restrict__ bc,
    unsigned short* __restrict__ hbf, float* __restrict__ cbuf) {
  __shared__ __align__(16) unsigned short Abuf[16 * KC];   // 6 KB
  __shared__ float zbuf[16 * 4 * HL];                      // 32 KB

  const int blk = blockIdx.x;                // 500 blocks of 16 rows
  const int tid = threadIdx.x;
  const int lane = tid & 31, w = tid >> 5;   // 8 waves
  const int lane16 = lane & 15, hi = lane >> 4;

  // stage A = [xt | h]: both source tiles are fully contiguous
  {
    const unsigned* xt32 = (const unsigned*)(xbf + ((size_t)t * BSn + (size_t)blk * 16) * HC);
    const unsigned* h32  = (const unsigned*)(hbf + (size_t)blk * 16 * HL);
    unsigned* A32 = (unsigned*)Abuf;               // 96 dwords per row
    for (int i = tid; i < 16 * HC / 2; i += 256) { // 512 dwords of xt
      int r = i >> 5, k = i & 31;
      A32[r * 96 + k] = xt32[i];
    }
    for (int i = tid; i < 16 * HL / 2; i += 256) { // 1024 dwords of h
      int r = i >> 6, k = i & 63;
      A32[r * 96 + 32 + k] = h32[i];
    }
  }
  __syncthreads();

  // pack 6 A fragments (16x32 each) from LDS
  FragBF Af[6];
#pragma unroll
  for (int kc = 0; kc < 6; ++kc) {
    const unsigned short* p = &Abuf[lane16 * KC + kc * 32 + hi * 8];
    Af[kc].q[0] = *(const uint4*)p;
    Af[kc].q[1] = *(const uint4*)(p + 16);
  }

  // each wave: 4 N-tiles of 16 columns
#pragma unroll
  for (int nt = 0; nt < 4; ++nt) {
    const int n0 = w * 64 + nt * 16;
    v8f acc = {};
#pragma unroll
    for (int kc = 0; kc < 6; ++kc) {
      // B fragment: column n0+lane16, K = kc*32 + hi*8 + {0..7, 16..23}
      const unsigned short* wp = Wc + (size_t)(n0 + lane16) * KC + kc * 32 + hi * 8;
      FragBF Bf;
      Bf.q[0] = *(const uint4*)wp;
      Bf.q[1] = *(const uint4*)(wp + 16);
      acc = __builtin_amdgcn_wmma_f32_16x16x32_bf16(false, Af[kc].v, false, Bf.v,
                                                    (short)0, acc, false, false);
    }
#pragma unroll
    for (int r = 0; r < 8; ++r) {
      int M = r + hi * 8;
      int n = n0 + lane16;
      zbuf[M * (4 * HL) + n] = acc[r] + bc[n];
    }
  }
  __syncthreads();

  // gates: i,f,g,o = z[:,0:128],[128:256],[256:384],[384:512]
#pragma unroll
  for (int p = 0; p < 8; ++p) {
    int ii = tid + p * 256;                 // 16*128 = 2048 work items
    int m = ii >> 7, j = ii & 127;
    int row = blk * 16 + m;
    float zi = zbuf[m * 512 + j];
    float zf = zbuf[m * 512 + 128 + j];
    float zg = zbuf[m * 512 + 256 + j];
    float zo = zbuf[m * 512 + 384 + j];
    float c = cbuf[(size_t)row * HL + j];
    float cn = sigm(zf) * c + sigm(zi) * tanhf(zg);
    float hn = sigm(zo) * tanhf(cn);
    cbuf[(size_t)row * HL + j] = cn;
    hbf[(size_t)row * HL + j] = f2bf(hn);
  }
}

// =====================================================================
// K5: output head  relu(h @ W_fc^T + b_fc)  -> [B,S,16]
// =====================================================================
__global__ __launch_bounds__(256) void k_fc(
    const unsigned short* __restrict__ hbf, const float* __restrict__ Wfc,
    const float* __restrict__ bfc, float* __restrict__ out) {
  int idx = blockIdx.x * blockDim.x + threadIdx.x;   // BS*16
  if (idx >= BSn * OUTn) return;
  int row = idx >> 4, o = idx & 15;
  const unsigned short* hp = hbf + (size_t)row * HL;
  const float* wp = Wfc + (size_t)o * HL;
  float acc = bfc[o];
#pragma unroll 8
  for (int j = 0; j < HL; ++j) acc += bf2f(hp[j]) * wp[j];
  out[idx] = fmaxf(acc, 0.0f);
}

// =====================================================================
extern "C" void kernel_launch(void* const* d_in, const int* in_sizes, int n_in,
                              void* d_out, int out_size, void* d_ws, size_t ws_size,
                              hipStream_t stream) {
  (void)in_sizes; (void)n_in; (void)out_size; (void)ws_size;
  const float* hist = (const float*)d_in[0];
  const float* stat = (const float*)d_in[1];
  const int*   ei   = (const int*)d_in[2];
  const float* Wg   = (const float*)d_in[3];
  const float* atts = (const float*)d_in[4];
  const float* attd = (const float*)d_in[5];
  const float* bg   = (const float*)d_in[6];
  const float* Wih  = (const float*)d_in[7];
  const float* Whh  = (const float*)d_in[8];
  const float* bih  = (const float*)d_in[9];
  const float* bhh  = (const float*)d_in[10];
  const float* Wfc  = (const float*)d_in[11];
  const float* bfc  = (const float*)d_in[12];
  float* out = (float*)d_out;

  char* ws = (char*)d_ws;
  float*          xw   = (float*)(ws + OFF_XW);
  float*          a_s  = (float*)(ws + OFF_AS);
  float*          a_d  = (float*)(ws + OFF_AD);
  unsigned short* xbf  = (unsigned short*)(ws + OFF_XBF);
  unsigned short* Wc   = (unsigned short*)(ws + OFF_WC);
  float*          bc   = (float*)(ws + OFF_BC);
  unsigned short* hbf  = (unsigned short*)(ws + OFF_HBF);
  float*          cbuf = (float*)(ws + OFF_CB);

  k_prep<<<1024, 256, 0, stream>>>(Wih, Whh, bih, bhh, Wc, bc, hbf, cbuf);

  // 6000 row-tiles, 4 waves per block
  k_gat_linear<<<1500, 128, 0, stream>>>(hist, stat, Wg, xw);

  k_att<<<(Gn * Sn * Hn + 255) / 256, 256, 0, stream>>>(xw, atts, attd, a_s, a_d);

  // one workgroup per graph; 288 KB dynamic LDS (m + den + agg)
  const size_t lds = (size_t)(2 * Sn * Hn * 4 + Sn * HC * 4);
  k_graph<<<Gn, 1024, lds, stream>>>(ei, a_s, a_d, xw, bg, xbf);

  for (int t = 0; t < Tn; ++t)
    k_lstm_step<<<BSn / 16, 256, 0, stream>>>(t, xbf, Wc, bc, hbf, cbuf);

  k_fc<<<(BSn * OUTn + 255) / 256, 256, 0, stream>>>(hbf, Wfc, bfc, out);
}
